// ConfigurableBackbone_48387101557374
// MI455X (gfx1250) — compile-verified, matrix operand order checked by reading, not run on hardware
//
#include <hip/hip_runtime.h>
#include <stdint.h>
#include <math.h>

typedef _Float16 half_t;
typedef __attribute__((ext_vector_type(16))) _Float16 v16h;
typedef __attribute__((ext_vector_type(8)))  float    v8f;

#define B_   2
#define CIN  3
#define LIN  8192
#define D_   128
#define H_   4
#define DH_  32
#define NL_  4
#define W_   64
#define DF_  512
#define S_   2048
#define M_   (B_*S_)      // 4096 rows
#define EPSF 1e-5f
#define NEGBIG (-1e30f)

// ------------------------------------------------------------------ conv1
__global__ __launch_bounds__(256)
void conv1_kernel(const float* __restrict__ x, const float* __restrict__ w,
                  const float* __restrict__ bias, float* __restrict__ out) {
  int idx = blockIdx.x * blockDim.x + threadIdx.x;   // [B, 64, 4096]
  const int L1 = LIN / 2;
  int o  = idx % L1;
  int co = (idx / L1) % (D_/2);
  int b  = idx / (L1 * (D_/2));
  float acc = bias[co];
  int base = 2*o - 7;
  for (int ci = 0; ci < CIN; ++ci) {
    const float* xp = x + (size_t)(b*CIN + ci)*LIN;
    const float* wp = w + (size_t)(co*CIN + ci)*15;
    #pragma unroll
    for (int k = 0; k < 15; ++k) {
      int p = base + k;
      float xv = (p >= 0 && p < LIN) ? xp[p] : 0.0f;
      acc = fmaf(xv, wp[k], acc);
    }
  }
  out[idx] = acc;
}

// ------------------------------------------------------------------ conv2
__global__ __launch_bounds__(256)
void conv2_kernel(const float* __restrict__ xin, const float* __restrict__ w,
                  const float* __restrict__ bias, float* __restrict__ out) {
  int idx = blockIdx.x * blockDim.x + threadIdx.x;   // [B, 128, 2048]
  int o  = idx % S_;
  int co = (idx / S_) % D_;
  int b  = idx / (S_ * D_);
  const int L1 = LIN / 2;
  float acc = bias[co];
  int base = 2*o - 1;
  const float* wp = w + (size_t)co * (D_/2) * 3;
  for (int ci = 0; ci < D_/2; ++ci) {
    const float* xp = xin + (size_t)(b*(D_/2) + ci)*L1;
    #pragma unroll
    for (int k = 0; k < 3; ++k) {
      int p = base + k;
      float xv = (p >= 0 && p < L1) ? xp[p] : 0.0f;
      acc = fmaf(xv, wp[ci*3 + k], acc);
    }
  }
  out[idx] = acc;
}

// -------------------------------------------------- batchnorm (training stats)
__global__ __launch_bounds__(256)
void bn_stats_kernel(const float* __restrict__ y, int C, int L,
                     float* __restrict__ stats) {
  int c = blockIdx.x;
  int tid = threadIdx.x;
  float s = 0.f, q = 0.f;
  int n = B_ * L;
  for (int i = tid; i < n; i += blockDim.x) {
    int b = i / L, pos = i % L;
    float v = y[(size_t)(b*C + c)*L + pos];
    s += v; q += v*v;
  }
  __shared__ float sh_s[256], sh_q[256];
  sh_s[tid] = s; sh_q[tid] = q; __syncthreads();
  for (int off = 128; off > 0; off >>= 1) {
    if (tid < off) { sh_s[tid] += sh_s[tid+off]; sh_q[tid] += sh_q[tid+off]; }
    __syncthreads();
  }
  if (tid == 0) {
    float mean = sh_s[0] / n;
    float var  = sh_q[0] / n - mean*mean;   // biased var (jnp.var ddof=0)
    stats[c]     = mean;
    stats[C + c] = rsqrtf(var + EPSF);
  }
}

__global__ __launch_bounds__(256)
void bn_relu_kernel(float* __restrict__ y, const float* __restrict__ stats,
                    const float* __restrict__ g, const float* __restrict__ bb,
                    int C, int L) {
  int idx = blockIdx.x * blockDim.x + threadIdx.x;
  int c = (idx / L) % C;
  float v = (y[idx] - stats[c]) * stats[C + c] * g[c] + bb[c];
  y[idx] = v > 0.f ? v : 0.f;
}

// ------------------------------- [B,C,S] -> [B,S,C], fp32 + f16 shadow copy
__global__ __launch_bounds__(256)
void transpose_kernel(const float* __restrict__ a, float* __restrict__ X,
                      half_t* __restrict__ Xh) {
  int idx = blockIdx.x * blockDim.x + threadIdx.x;   // output-linear
  int c = idx % D_;
  int s = (idx / D_) % S_;
  int b = idx / (D_ * S_);
  float v = a[(size_t)(b*D_ + c)*S_ + s];
  X[idx]  = v;
  Xh[idx] = (half_t)v;
}

// ---------------------- pack fp32 weights [K,N] into WMMA B-fragment layout
// layout: tile t = nt*(K/32)+kt holds 512 halves: lane l (32B contiguous):
//   n = nt*16 + (l&15);  half h -> k = kt*32 + (h>>3)*16 + (l>>4)*8 + (h&7)
__global__ __launch_bounds__(256)
void pack_w_kernel(const float* __restrict__ W, half_t* __restrict__ dst,
                   int K, int N) {
  int idx = blockIdx.x * blockDim.x + threadIdx.x;   // over K*N halves
  int h = idx & 15;
  int l = (idx >> 4) & 31;
  int t = idx >> 9;
  int KT = K >> 5;
  int kt = t % KT;
  int nt = t / KT;
  int n = nt*16 + (l & 15);
  int k = kt*32 + ((h >> 3) << 4) + ((l >> 4) << 3) + (h & 7);
  dst[idx] = (half_t)W[(size_t)k*N + n];
}

// ------------------------------------------------------------ WMMA GEMM
// Out[M,N] = A[M,K](f16) @ Wp(f16, pre-packed fragments) + bias (+relu|+resid)
// Block: 256 thr = 8 waves, tile 64(M) x 64(N), BK=32.
// A-tiles double-buffered in LDS via global_load_async_to_lds_b128 (ASYNCcnt);
// B fragments are single 32B coalesced global loads (L2-resident), prefetched.
template<int EPI>   // 0: bias->f32   1: bias+relu->f16   2: bias+residual->f32
__global__ __launch_bounds__(256)
void gemm_wmma(const half_t* __restrict__ A, const half_t* __restrict__ Wp,
               const float* __restrict__ bias, const float* __restrict__ Res,
               float* __restrict__ OutF, half_t* __restrict__ OutH,
               int N, int K) {
  const int SK = 40;                          // padded k-stride (halves)
  __shared__ half_t As[2][64 * SK];           // double-buffered A tiles
  const int tid  = threadIdx.x;
  const int lane = tid & 31;
  const int wv   = tid >> 5;
  const int tc   = wv & 3;
  const int tr0  = (wv >> 2) * 2;
  const int m0 = blockIdx.y * 64;
  const int KT = K >> 5;

  // async staging coords: one 16B chunk per thread per tile
  const int srow = tid >> 2, sch = tid & 3;
  const half_t* gA = A + (size_t)(m0 + srow) * K + sch * 8;

  { // stage tile 0
    uint32_t lds = (uint32_t)(size_t)(&As[0][srow*SK + sch*8]);
    asm volatile("global_load_async_to_lds_b128 %0, %1, off"
                 :: "v"(lds), "v"((unsigned long long)(size_t)gA) : "memory");
  }

  v8f c0 = {}, c1 = {};
  const int l15  = lane & 15;
  const int klo8 = (lane >> 4) * 8;           // ISA 16-bit fragment lane->K map
  const int mrowA0 = tr0*16 + l15;
  const int mrowA1 = mrowA0 + 16;
  const int ntile  = blockIdx.x * 4 + tc;
  const half_t* bp = Wp + (size_t)ntile * KT * 512 + lane * 16;

  for (int kt = 0; kt < KT; ++kt) {
    const int cur = kt & 1;
    if (kt + 1 < KT) {
      uint32_t lds = (uint32_t)(size_t)(&As[cur ^ 1][srow*SK + sch*8]);
      asm volatile("global_load_async_to_lds_b128 %0, %1, off"
                   :: "v"(lds),
                      "v"((unsigned long long)(size_t)(gA + (kt + 1) * 32))
                   : "memory");
      __builtin_prefetch(bp + (size_t)(kt + 1) * 512, 0, 0);
      asm volatile("s_wait_asynccnt 0x1" ::: "memory");   // tile kt resident
    } else {
      asm volatile("s_wait_asynccnt 0x0" ::: "memory");
    }
    __syncthreads();

    v16h bv = *(const v16h*)(bp + (size_t)kt * 512);      // B fragment (32B)
    union U { uint32_t u[8]; v16h v; } a0u, a1u;
    const half_t* as = &As[cur][0];
    #pragma unroll
    for (int j = 0; j < 8; ++j) {
      int k = ((j >> 2) << 4) + klo8 + ((j & 3) << 1);
      a0u.u[j] = *(const uint32_t*)(as + mrowA0*SK + k);
      a1u.u[j] = *(const uint32_t*)(as + mrowA1*SK + k);
    }
    c0 = __builtin_amdgcn_wmma_f32_16x16x32_f16(false, a0u.v, false, bv,
                                                (short)0, c0, false, false);
    c1 = __builtin_amdgcn_wmma_f32_16x16x32_f16(false, a1u.v, false, bv,
                                                (short)0, c1, false, false);
    __syncthreads();   // all waves done with As[cur] before it is restaged
  }

  const int nn = blockIdx.x*64 + tc*16 + l15;
  const float bvv = bias[nn];
  #pragma unroll
  for (int t = 0; t < 2; ++t) {
    v8f c = t ? c1 : c0;
    int mbase = m0 + (tr0 + t)*16 + ((lane >> 4) * 8);    // C/D row map
    #pragma unroll
    for (int r = 0; r < 8; ++r) {
      int base = (mbase + r) * N + nn;                    // 32-bit index math
      float val = c[r] + bvv;
      if (EPI == 1)      { val = val > 0.f ? val : 0.f; OutH[base] = (half_t)val; }
      else if (EPI == 2) { OutF[base] = val + Res[base]; }
      else               { OutF[base] = val; }
    }
  }
}

// ------------------------------------------------- banded attention (wave/query)
__global__ __launch_bounds__(256)
void attn_kernel(const float* __restrict__ Q, const float* __restrict__ Kb,
                 const float* __restrict__ V, half_t* __restrict__ O) {
  const int lane = threadIdx.x & 31;
  const int wid  = blockIdx.x * (blockDim.x >> 5) + (threadIdx.x >> 5);
  int s = wid % S_;
  int h = (wid / S_) % H_;
  int b = wid / (S_ * H_);
  const int m = b*S_ + s;
  const float qv = Q[(size_t)m*D_ + h*DH_ + lane];   // lane holds q[d=lane]
  const int jlo = max(0, s - W_);
  const int jhi = min(S_ - 1, s + W_);
  const int cnt   = jhi - jlo + 1;                   // 65..129
  const int nslot = (cnt + 31) >> 5;                 // 3..5
  const float scale = 0.17677669529663687f;          // 1/sqrt(32)

  float e[5];
  float mx = NEGBIG;
  #pragma unroll
  for (int slot = 0; slot < 5; ++slot) {
    float sc = NEGBIG;
    if (slot < nslot) {
      int j = jlo + slot*32 + lane;
      bool ok = j <= jhi;
      int jj = ok ? j : jhi;
      const float* kp = Kb + (size_t)(b*S_ + jj)*D_ + h*DH_;
      float acc = 0.f;
      #pragma unroll
      for (int d = 0; d < 32; ++d)
        acc = fmaf(__shfl(qv, d, 32), kp[d], acc);
      sc = ok ? acc * scale : NEGBIG;
    }
    e[slot] = sc;
    mx = fmaxf(mx, sc);
  }
  #pragma unroll
  for (int off = 16; off >= 1; off >>= 1) mx = fmaxf(mx, __shfl_xor(mx, off, 32));

  float denom = 0.f;
  #pragma unroll
  for (int slot = 0; slot < 5; ++slot) {
    float w = (e[slot] <= NEGBIG) ? 0.f : __expf(e[slot] - mx);
    e[slot] = w;
    denom += w;
  }
  #pragma unroll
  for (int off = 16; off >= 1; off >>= 1) denom += __shfl_xor(denom, off, 32);

  float o = 0.f;
  #pragma unroll
  for (int slot = 0; slot < 5; ++slot) {
    if (slot >= nslot) break;
    float es  = e[slot];
    int jbase = jlo + slot*32;
    int tmax  = min(32, jhi - jbase + 1);
    for (int t = 0; t < tmax; ++t) {
      float w = __shfl(es, t, 32);
      o = fmaf(w, V[(size_t)(b*S_ + jbase + t)*D_ + h*DH_ + lane], o);
    }
  }
  O[(size_t)m*D_ + h*DH_ + lane] = (half_t)(o / denom);
}

// ----------------------- LayerNorm (wave/row): fp32 out + f16 shadow copy
__global__ __launch_bounds__(256)
void ln_kernel(const float* __restrict__ Y, const float* __restrict__ g,
               const float* __restrict__ bb, float* __restrict__ Xo,
               half_t* __restrict__ Xh) {
  const int lane = threadIdx.x & 31;
  const int row  = blockIdx.x * (blockDim.x >> 5) + (threadIdx.x >> 5);
  const float* y = Y + (size_t)row * D_;
  float v0 = y[lane], v1 = y[lane+32], v2 = y[lane+64], v3 = y[lane+96];
  float s = v0+v1+v2+v3;
  float q = v0*v0 + v1*v1 + v2*v2 + v3*v3;
  #pragma unroll
  for (int off = 16; off >= 1; off >>= 1) {
    s += __shfl_xor(s, off, 32);
    q += __shfl_xor(q, off, 32);
  }
  float mean = s * (1.0f/D_);
  float var  = q * (1.0f/D_) - mean*mean;
  float r = rsqrtf(var + EPSF);
  float* xo = Xo + (size_t)row * D_;
  half_t* xh = Xh + (size_t)row * D_;
  float o0 = (v0-mean)*r*g[lane]    + bb[lane];
  float o1 = (v1-mean)*r*g[lane+32] + bb[lane+32];
  float o2 = (v2-mean)*r*g[lane+64] + bb[lane+64];
  float o3 = (v3-mean)*r*g[lane+96] + bb[lane+96];
  xo[lane]    = o0;  xh[lane]    = (half_t)o0;
  xo[lane+32] = o1;  xh[lane+32] = (half_t)o1;
  xo[lane+64] = o2;  xh[lane+64] = (half_t)o2;
  xo[lane+96] = o3;  xh[lane+96] = (half_t)o3;
}

// ================================================================== launcher
extern "C" void kernel_launch(void* const* d_in, const int* in_sizes, int n_in,
                              void* d_out, int out_size, void* d_ws, size_t ws_size,
                              hipStream_t stream) {
  const float* x    = (const float*)d_in[0];
  const float* c1w  = (const float*)d_in[1];
  const float* c1b  = (const float*)d_in[2];
  const float* bn1g = (const float*)d_in[3];
  const float* bn1b = (const float*)d_in[4];
  const float* c2w  = (const float*)d_in[5];
  const float* c2b  = (const float*)d_in[6];
  const float* bn2g = (const float*)d_in[7];
  const float* bn2b = (const float*)d_in[8];
  const float* wq = (const float*)d_in[9];   const float* bq = (const float*)d_in[10];
  const float* wk = (const float*)d_in[11];  const float* bk = (const float*)d_in[12];
  const float* wv = (const float*)d_in[13];  const float* bv = (const float*)d_in[14];
  const float* wo = (const float*)d_in[15];  const float* bo = (const float*)d_in[16];
  const float* lng = (const float*)d_in[17]; const float* lnb = (const float*)d_in[18];
  const float* w1 = (const float*)d_in[19];  const float* b1 = (const float*)d_in[20];
  const float* w2 = (const float*)d_in[21];  const float* b2 = (const float*)d_in[22];
  (void)in_sizes; (void)n_in; (void)out_size; (void)ws_size;

  char* wsb = (char*)d_ws;
  auto carve = [&](size_t bytes) -> char* {
    char* p = wsb; wsb += (bytes + 255) & ~(size_t)255; return p;
  };
  float*  a1    = (float*)carve((size_t)B_*(D_/2)*(LIN/2)*4);
  float*  a2    = (float*)carve((size_t)B_*D_*S_*4);
  float*  X     = (float*)carve((size_t)M_*D_*4);
  half_t* Xh    = (half_t*)carve((size_t)M_*D_*2);
  float*  Qb    = (float*)carve((size_t)M_*D_*4);
  float*  Kbf   = (float*)carve((size_t)M_*D_*4);
  float*  Vb    = (float*)carve((size_t)M_*D_*4);
  half_t* Oh    = (half_t*)carve((size_t)M_*D_*2);
  float*  Yb    = (float*)carve((size_t)M_*D_*4);
  half_t* Hh    = (half_t*)carve((size_t)M_*DF_*2);
  half_t* Wpack = (half_t*)carve((size_t)NL_*(4*D_*D_ + 2*D_*DF_)*2);
  float*  stats = (float*)carve(256*4);

  const int TB = 256;
  const size_t LW = 4*D_*D_ + 2*D_*DF_;   // packed halves per layer (196608)

  // one-time weight packing into WMMA fragment layout (f16)
  for (int i = 0; i < NL_; ++i) {
    half_t* wp = Wpack + i*LW;
    pack_w_kernel<<<(D_*D_)/TB,  TB, 0, stream>>>(wq + (size_t)i*D_*D_,  wp,            D_,  D_);
    pack_w_kernel<<<(D_*D_)/TB,  TB, 0, stream>>>(wk + (size_t)i*D_*D_,  wp +   D_*D_,  D_,  D_);
    pack_w_kernel<<<(D_*D_)/TB,  TB, 0, stream>>>(wv + (size_t)i*D_*D_,  wp + 2*D_*D_,  D_,  D_);
    pack_w_kernel<<<(D_*D_)/TB,  TB, 0, stream>>>(wo + (size_t)i*D_*D_,  wp + 3*D_*D_,  D_,  D_);
    pack_w_kernel<<<(D_*DF_)/TB, TB, 0, stream>>>(w1 + (size_t)i*D_*DF_, wp + 4*D_*D_,  D_,  DF_);
    pack_w_kernel<<<(DF_*D_)/TB, TB, 0, stream>>>(w2 + (size_t)i*DF_*D_, wp + 4*D_*D_ + D_*DF_, DF_, D_);
  }

  // conv stem
  conv1_kernel<<<(B_*(D_/2)*(LIN/2))/TB, TB, 0, stream>>>(x, c1w, c1b, a1);
  bn_stats_kernel<<<D_/2, TB, 0, stream>>>(a1, D_/2, LIN/2, stats);
  bn_relu_kernel<<<(B_*(D_/2)*(LIN/2))/TB, TB, 0, stream>>>(a1, stats, bn1g, bn1b, D_/2, LIN/2);
  conv2_kernel<<<(B_*D_*S_)/TB, TB, 0, stream>>>(a1, c2w, c2b, a2);
  bn_stats_kernel<<<D_, TB, 0, stream>>>(a2, D_, S_, stats);
  bn_relu_kernel<<<(B_*D_*S_)/TB, TB, 0, stream>>>(a2, stats, bn2g, bn2b, D_, S_);
  transpose_kernel<<<(B_*S_*D_)/TB, TB, 0, stream>>>(a2, X, Xh);

  dim3 gproj(D_/64, M_/64);    // (2, 64)
  dim3 gff1(DF_/64, M_/64);    // (8, 64)
  for (int i = 0; i < NL_; ++i) {
    const half_t* wpq = Wpack + i*LW;
    const half_t* wpk = wpq +   D_*D_;
    const half_t* wpv = wpq + 2*D_*D_;
    const half_t* wpo = wpq + 3*D_*D_;
    const half_t* wp1 = wpq + 4*D_*D_;
    const half_t* wp2 = wpq + 4*D_*D_ + D_*DF_;
    const float* bqi = bq + (size_t)i*D_;   const float* bki = bk + (size_t)i*D_;
    const float* bvi = bv + (size_t)i*D_;   const float* boi = bo + (size_t)i*D_;
    const float* b1i = b1 + (size_t)i*DF_;  const float* b2i = b2 + (size_t)i*D_;
    const float* lgi = lng + (size_t)i*D_;  const float* lbi = lnb + (size_t)i*D_;

    gemm_wmma<0><<<gproj, TB, 0, stream>>>(Xh, wpq, bqi, nullptr, Qb,  nullptr, D_, D_);
    gemm_wmma<0><<<gproj, TB, 0, stream>>>(Xh, wpk, bki, nullptr, Kbf, nullptr, D_, D_);
    gemm_wmma<0><<<gproj, TB, 0, stream>>>(Xh, wpv, bvi, nullptr, Vb,  nullptr, D_, D_);
    attn_kernel<<<(B_*H_*S_)/8, TB, 0, stream>>>(Qb, Kbf, Vb, Oh);
    gemm_wmma<2><<<gproj, TB, 0, stream>>>(Oh, wpo, boi, X, Yb, nullptr, D_, D_); // +residual
    ln_kernel<<<M_/8, TB, 0, stream>>>(Yb, lgi, lbi, X, Xh);
    gemm_wmma<1><<<gff1,  TB, 0, stream>>>(Xh, wp1, b1i, nullptr, nullptr, Hh, DF_, D_); // +relu
    gemm_wmma<2><<<gproj, TB, 0, stream>>>(Hh, wp2, b2i, X, Yb, nullptr, D_, DF_);       // +residual
    float* dst = (i == NL_ - 1) ? (float*)d_out : X;
    ln_kernel<<<M_/8, TB, 0, stream>>>(Yb, lgi, lbi, dst, Xh);
  }
}
